// INE_Estimator_42099269435842
// MI455X (gfx1250) — matrix operand order repaired and checked by reading, not compiled, not run on hardware
//
#include <hip/hip_runtime.h>
#include <hip/hip_bf16.h>
#include <math.h>

// Problem sizes (fixed by the reference)
constexpr int NN = 8192;   // nodes
constexpr int HH = 256;    // hidden width

typedef __attribute__((ext_vector_type(16))) __bf16 v16bf;
typedef __attribute__((ext_vector_type(8)))  __bf16 v8bf;
typedef __attribute__((ext_vector_type(8)))  float  v8f;
typedef __attribute__((ext_vector_type(4)))  float  v4f;   // clang vector: OK for NT builtins

// ---------------------------------------------------------------------------
// Kernel 1: per-row degree stats.
//   degraw[i] = sum_j A[i,j]
//   dinv[i]   = 1/sqrt(degraw+1)            (A_hat = A + I degrees)
//   expo[i]   = t[i]*degraw/(degraw+1e-8)   (faithful to the broadcast in ref)
// ---------------------------------------------------------------------------
__global__ __launch_bounds__(256)
void rowstats_kernel(const float* __restrict__ A,
                     const float* __restrict__ treat,
                     float* __restrict__ dinv,
                     float* __restrict__ expo) {
  __shared__ float red[256];
  const int i = blockIdx.x;
  const int j = threadIdx.x;
  const v4f* row = (const v4f*)(A + (size_t)i * NN);
  float s = 0.0f;
  // NN/4 = 2048 vec4 per row; 256 threads -> 8 vec loads each (streaming, NT)
  for (int k = j; k < NN / 4; k += 256) {
    v4f v = __builtin_nontemporal_load(row + k);
    s += v[0] + v[1] + v[2] + v[3];
  }
  red[j] = s;
  __syncthreads();
  for (int ofs = 128; ofs > 0; ofs >>= 1) {
    if (j < ofs) red[j] += red[j + ofs];
    __syncthreads();
  }
  if (j == 0) {
    float d = red[0];
    dinv[i] = rsqrtf(d + 1.0f);
    expo[i] = treat[i] * d / (d + 1e-8f);
  }
}

// ---------------------------------------------------------------------------
// Kernel 2: small GEMM  X = In[N,K] @ W[K,H]  (fp32, exact), fused with the
// production of the WMMA B-operand:  Ys = dinv .* X  split into hi/lo bf16,
// stored TRANSPOSED ([H][N], n-major) so that WMMA B-fragments are two
// contiguous 16B loads per lane.
// One block per row i, one thread per output column j.
// ---------------------------------------------------------------------------
__global__ __launch_bounds__(256)
void gemm_split_kernel(const float* __restrict__ In,   // [N][K]
                       const float* __restrict__ W,    // [K][H]
                       const float* __restrict__ dinv, // [N]
                       float* __restrict__ X,          // [N][H]  raw product
                       __bf16* __restrict__ XT_hi,     // [H][N]
                       __bf16* __restrict__ XT_lo,     // [H][N]
                       int K) {
  __shared__ float srow[256];
  const int i = blockIdx.x;
  const int j = threadIdx.x;
  if (j < K) srow[j] = In[(size_t)i * K + j];
  __syncthreads();
  float s = 0.0f;
  for (int k = 0; k < K; ++k) s = fmaf(srow[k], W[(size_t)k * HH + j], s);
  X[(size_t)i * HH + j] = s;
  float ys = dinv[i] * s;
  __bf16 hi = (__bf16)ys;
  __bf16 lo = (__bf16)(ys - (float)hi);
  XT_hi[(size_t)j * NN + i] = hi;
  XT_lo[(size_t)j * NN + i] = lo;
}

// ---------------------------------------------------------------------------
// Kernel 3: the heavy propagation  out = relu( Dinv (A+I) Dinv X + b )
//   acc = A @ (Dinv X)  via v_wmma_f32_16x16x32_bf16, split-bf16 operand
//   out[i,j] = relu( dinv_i*acc + dinv_i^2 * X[i,j] + b_j )
// A (0/1 fp32) is streamed NT and converted to bf16 in-register (exact).
// 4 waves / block; wave w owns rows [blk*64 + w*16, +16) x all 256 cols.
// B-fragments use a depth-4 software-pipelined register ring so WMMAs never
// wait on a just-issued L2 load.
// ---------------------------------------------------------------------------
__device__ inline v16bf pack_a_frag(v4f a, v4f b, v4f c, v4f d) {
  v16bf r;
  r[0]  = (__bf16)a[0]; r[1]  = (__bf16)a[1]; r[2]  = (__bf16)a[2]; r[3]  = (__bf16)a[3];
  r[4]  = (__bf16)b[0]; r[5]  = (__bf16)b[1]; r[6]  = (__bf16)b[2]; r[7]  = (__bf16)b[3];
  r[8]  = (__bf16)c[0]; r[9]  = (__bf16)c[1]; r[10] = (__bf16)c[2]; r[11] = (__bf16)c[3];
  r[12] = (__bf16)d[0]; r[13] = (__bf16)d[1]; r[14] = (__bf16)d[2]; r[15] = (__bf16)d[3];
  return r;
}

__device__ inline v16bf load_b_frag(const __bf16* p) {
  v8bf lo = *(const v8bf*)p;        // K offsets 0..7   -> VGPRs 0..3
  v8bf hi = *(const v8bf*)(p + 16); // K offsets 16..23 -> VGPRs 4..7
  v16bf r;
#pragma unroll
  for (int e = 0; e < 8; ++e) { r[e] = lo[e]; r[8 + e] = hi[e]; }
  return r;
}

__global__ __launch_bounds__(128)
void prop_kernel(const float*  __restrict__ A,      // [N][N] fp32 0/1
                 const __bf16* __restrict__ BT_hi,  // [H][N]
                 const __bf16* __restrict__ BT_lo,  // [H][N]
                 const float*  __restrict__ X,      // [N][H] pre-aggregation
                 const float*  __restrict__ dinv,   // [N]
                 const float*  __restrict__ bias,   // [H]
                 float* __restrict__ out) {         // [N][H]
  const int wave = threadIdx.x >> 5;
  const int lane = threadIdx.x & 31;
  const int half = lane >> 4;   // K-half selector for A/B fragments
  const int l16  = lane & 15;
  const int m0   = blockIdx.x * 64 + wave * 16;   // first row of this wave
  const int arow = m0 + l16;                      // row this lane loads for A

  v8f acc[16];
#pragma unroll
  for (int n = 0; n < 16; ++n)
#pragma unroll
    for (int g = 0; g < 8; ++g) acc[n][g] = 0.0f;

  const float* Arow = A + (size_t)arow * NN;
  // per-lane fragment bases
  const __bf16* bh_base = BT_hi + (size_t)l16 * NN + (size_t)(half * 8);
  const __bf16* bl_base = BT_lo + (size_t)l16 * NN + (size_t)(half * 8);
  // fragment offset for (k0, n): n*16*NN + k0
  auto boff = [&](int k0, int n) -> size_t {
    return (size_t)n * (16 * NN) + (size_t)k0;
  };

  // ---- prologue: prefetch A for k0=0 and B-fragments for iterations 0..3
  const v4f* pa0 = (const v4f*)(Arow + 0 + half * 8);
  const v4f* pa1 = (const v4f*)(Arow + 16 + half * 8);
  v4f a0 = __builtin_nontemporal_load(pa0 + 0);
  v4f a1 = __builtin_nontemporal_load(pa0 + 1);
  v4f a2 = __builtin_nontemporal_load(pa1 + 0);
  v4f a3 = __builtin_nontemporal_load(pa1 + 1);

  v16bf bh_ring[4], bl_ring[4];
#pragma unroll
  for (int n = 0; n < 4; ++n) {
    bh_ring[n] = load_b_frag(bh_base + boff(0, n));
    bl_ring[n] = load_b_frag(bl_base + boff(0, n));
  }

  for (int k0 = 0; k0 < NN; k0 += 32) {
    v16bf af = pack_a_frag(a0, a1, a2, a3);

    // prefetch the A fragment of the NEXT K-step (wraps harmlessly at end)
    const int kn = (k0 + 32) & (NN - 1);
    const v4f* qa0 = (const v4f*)(Arow + kn + half * 8);
    const v4f* qa1 = (const v4f*)(Arow + kn + 16 + half * 8);
    a0 = __builtin_nontemporal_load(qa0 + 0);
    a1 = __builtin_nontemporal_load(qa0 + 1);
    a2 = __builtin_nontemporal_load(qa1 + 0);
    a3 = __builtin_nontemporal_load(qa1 + 1);

#pragma unroll
    for (int n = 0; n < 16; ++n) {
      const int slot = n & 3;
      acc[n] = __builtin_amdgcn_wmma_f32_16x16x32_bf16(
          false, af, false, bh_ring[slot], (short)0, acc[n], false, false);
      acc[n] = __builtin_amdgcn_wmma_f32_16x16x32_bf16(
          false, af, false, bl_ring[slot], (short)0, acc[n], false, false);
      // refill this slot with the fragment 4 iterations ahead
      int n4 = n + 4;
      int kp = k0;
      if (n4 >= 16) { n4 -= 16; kp = kn; }   // compile-time per unrolled n
      bh_ring[slot] = load_b_frag(bh_base + boff(kp, n4));
      bl_ring[slot] = load_b_frag(bl_base + boff(kp, n4));
    }
  }

  // Epilogue: C layout -> row = m0 + g + 8*half, col = n*16 + l16
  float di[8];
#pragma unroll
  for (int g = 0; g < 8; ++g) di[g] = dinv[m0 + g + 8 * half];

#pragma unroll
  for (int n = 0; n < 16; ++n) {
    const int col = n * 16 + l16;
    const float b = bias[col];
#pragma unroll
    for (int g = 0; g < 8; ++g) {
      const int row = m0 + g + 8 * half;
      const float d = di[g];
      float v = d * acc[n][g] + d * d * X[(size_t)row * HH + col] + b;
      out[(size_t)row * HH + col] = fmaxf(v, 0.0f);
    }
  }
}

// ---------------------------------------------------------------------------
// Kernel 4: final stage, one block per row i:
//   emb[i,:] = relu( h2[i,:] @ W_emb[0:256,:] + expo[i]*W_emb[256,:] + b_emb )
//   y0[i] = emb[i,:].W_t0 + b_t0 ; y1[i] = emb[i,:].W_t1 + b_t1
// ---------------------------------------------------------------------------
__global__ __launch_bounds__(256)
void final_kernel(const float* __restrict__ Hin,    // [N][256]
                  const float* __restrict__ W_emb,  // [257][256]
                  const float* __restrict__ b_emb,  // [256]
                  const float* __restrict__ expo,   // [N]
                  const float* __restrict__ W_t0, const float* __restrict__ b_t0,
                  const float* __restrict__ W_t1, const float* __restrict__ b_t1,
                  float* __restrict__ y0, float* __restrict__ y1,
                  float* __restrict__ emb) {
  __shared__ float srow[256];
  __shared__ float red0[256];
  __shared__ float red1[256];
  const int i = blockIdx.x;
  const int j = threadIdx.x;
  srow[j] = Hin[(size_t)i * HH + j];
  __syncthreads();
  float s = 0.0f;
  for (int k = 0; k < HH; ++k) s = fmaf(srow[k], W_emb[(size_t)k * HH + j], s);
  s += expo[i] * W_emb[(size_t)HH * HH + j] + b_emb[j];
  float e = fmaxf(s, 0.0f);
  emb[(size_t)i * HH + j] = e;
  red0[j] = e * W_t0[j];
  red1[j] = e * W_t1[j];
  __syncthreads();
  for (int ofs = 128; ofs > 0; ofs >>= 1) {
    if (j < ofs) { red0[j] += red0[j + ofs]; red1[j] += red1[j + ofs]; }
    __syncthreads();
  }
  if (j == 0) {
    y0[i] = red0[0] + b_t0[0];
    y1[i] = red1[0] + b_t1[0];
  }
}

// ---------------------------------------------------------------------------
// Launcher
// ---------------------------------------------------------------------------
extern "C" void kernel_launch(void* const* d_in, const int* in_sizes, int n_in,
                              void* d_out, int out_size, void* d_ws, size_t ws_size,
                              hipStream_t stream) {
  (void)in_sizes; (void)n_in; (void)out_size; (void)ws_size;

  const float* A     = (const float*)d_in[0];
  const float* feat  = (const float*)d_in[1];
  const float* treat = (const float*)d_in[2];
  const float* W1    = (const float*)d_in[3];
  const float* b1    = (const float*)d_in[4];
  const float* W2    = (const float*)d_in[5];
  const float* b2    = (const float*)d_in[6];
  const float* Wemb  = (const float*)d_in[7];
  const float* bemb  = (const float*)d_in[8];
  const float* Wt0   = (const float*)d_in[9];
  const float* bt0   = (const float*)d_in[10];
  const float* Wt1   = (const float*)d_in[11];
  const float* bt1   = (const float*)d_in[12];

  float* out = (float*)d_out;
  float* y0  = out;            // [N]
  float* y1  = out + NN;       // [N]
  float* emb = out + 2 * NN;   // [N][H]

  // Workspace carve-up (~24.3 MB, 256B-aligned chunks)
  char* ws = (char*)d_ws;
  size_t off = 0;
  auto carve = [&](size_t bytes) {
    void* p = ws + off;
    off += (bytes + 255) & ~(size_t)255;
    return p;
  };
  float*  dinv  = (float*)carve((size_t)NN * sizeof(float));
  float*  expo  = (float*)carve((size_t)NN * sizeof(float));
  float*  X     = (float*)carve((size_t)NN * HH * sizeof(float));     // 8 MB
  float*  h     = (float*)carve((size_t)NN * HH * sizeof(float));     // 8 MB
  __bf16* XT_hi = (__bf16*)carve((size_t)HH * NN * sizeof(__bf16));   // 4 MB
  __bf16* XT_lo = (__bf16*)carve((size_t)HH * NN * sizeof(__bf16));   // 4 MB

  // 1) degrees / normalization / exposure
  rowstats_kernel<<<NN, 256, 0, stream>>>(A, treat, dinv, expo);

  // 2) X1 = feat @ W1 ; split/transpose Dinv*X1
  gemm_split_kernel<<<NN, 256, 0, stream>>>(feat, W1, dinv, X, XT_hi, XT_lo, 128);

  // 3) h1 = relu(nA @ X1 + b1)   (WMMA)
  prop_kernel<<<NN / 64, 128, 0, stream>>>(A, XT_hi, XT_lo, X, dinv, b1, h);

  // 4) X2 = h1 @ W2 ; split/transpose Dinv*X2   (X1 buffers dead -> reuse)
  gemm_split_kernel<<<NN, 256, 0, stream>>>(h, W2, dinv, X, XT_hi, XT_lo, 256);

  // 5) h2 = relu(nA @ X2 + b2)   (WMMA; h1 dead -> reuse h)
  prop_kernel<<<NN / 64, 128, 0, stream>>>(A, XT_hi, XT_lo, X, dinv, b2, h);

  // 6) emb / y0 / y1
  final_kernel<<<NN, 256, 0, stream>>>(h, Wemb, bemb, expo,
                                       Wt0, bt0, Wt1, bt1, y0, y1, emb);
}
